// PSN_89764816487053
// MI455X (gfx1250) — compile-verified, matrix operand order checked by reading, not run on hardware
//
#include <hip/hip_runtime.h>
#include <hip/hip_bf16.h>

typedef __attribute__((ext_vector_type(16))) _Float16 v16h;
typedef __attribute__((ext_vector_type(8)))  _Float16 v8h;
typedef __attribute__((ext_vector_type(8)))  float    v8f;
typedef __attribute__((ext_vector_type(4)))  unsigned int u32x4;
typedef __attribute__((ext_vector_type(8)))  int          i32x8;
typedef __attribute__((ext_vector_type(4)))  int          i32x4;

#define NB    32
#define EB    256
#define CDIM  256
#define KCB   8192
#define NTOT  (NB * EB)          // 8192 rows
#define NTILE 64                 // rows per workgroup
#define KTILE 32                 // codebook entries per chunk
#define NCHUNK (KCB / KTILE)     // 256
#define ASTRIDE 264              // padded LDS row stride in halfs (528 B = 132 dwords -> conflict-free)
#define A_BYTES (NTILE * ASTRIDE * 2)   // 33792
#define BBUF_BYTES (KTILE * ASTRIDE * 2) // 16896 per B buffer
#define SMEM_TOTAL (A_BYTES + 2 * BBUF_BYTES) // 67584 (< 320 KB WGP pool; dynamic LDS)
#define BETA  0.25f

// ---------------------------------------------------------------------------
// Kernel 0: zero the scalar loss accumulator (graph-capture safe init)
// ---------------------------------------------------------------------------
__global__ void vq_init(float* __restrict__ loss) {
    if (threadIdx.x == 0 && blockIdx.x == 0) *loss = 0.0f;
}

// ---------------------------------------------------------------------------
// Kernel 1: f32 -> f16 conversion of latents + codebook, codebook row norms.
// ---------------------------------------------------------------------------
__global__ __launch_bounds__(256) void vq_prep(const float* __restrict__ lat,
                                               const float* __restrict__ cb,
                                               _Float16* __restrict__ lat_h,
                                               _Float16* __restrict__ cb_h,
                                               float* __restrict__ cnorm) {
    __shared__ float ws[8];
    const int bid = blockIdx.x;
    const int tid = threadIdx.x;
    if (bid < KCB) {
        const float v = cb[(size_t)bid * CDIM + tid];
        cb_h[(size_t)bid * CDIM + tid] = (_Float16)v;
        float s = v * v;
        #pragma unroll
        for (int off = 16; off > 0; off >>= 1) s += __shfl_xor(s, off, 32);
        if ((tid & 31) == 0) ws[tid >> 5] = s;
        __syncthreads();
        if (tid == 0) {
            float t = 0.0f;
            #pragma unroll
            for (int i = 0; i < 8; ++i) t += ws[i];
            cnorm[bid] = t;
        }
    } else {
        const int r = bid - KCB;
        lat_h[(size_t)r * CDIM + tid] = (_Float16)lat[(size_t)r * CDIM + tid];
    }
}

__device__ __forceinline__ v16h ld_frag(const _Float16* p) {
    // 528-byte row stride guarantees 16B alignment; build v16h from two v8h.
    v8h lo = *(const v8h*)p;
    v8h hi = *(const v8h*)(p + 8);
    return __builtin_shufflevector(lo, hi, 0, 1, 2, 3, 4, 5, 6, 7,
                                           8, 9, 10, 11, 12, 13, 14, 15);
}

// ---------------------------------------------------------------------------
// TDM descriptor: DMA one 32x256-f16 codebook tile into LDS, inserting
// 4 dwords (16 B) of padding after every 128 dwords (512 B = one 256-half
// row) -> padded row stride 528 B == ASTRIDE halfs, matching ld_frag.
//   group0: count=1 | lds_addr | global_addr(57b) | type=2
//   group1: data_size=2B, pad_enable, pad_interval=6 (128 dw), pad_amount=3
//           (4 dw), tensor_dim0=256, tensor_dim1=8192, tile=256x32,
//           tensor_dim0_stride=256
//   groups 2/3: zero (2-D tensor)
// ---------------------------------------------------------------------------
__device__ __forceinline__ void tdm_load_tile(const _Float16* gsrc, unsigned int lds_off) {
    const unsigned long long ga = (unsigned long long)(uintptr_t)gsrc;
    u32x4 g0 = { 1u,                                   // count=1 (valid user D#)
                 lds_off,                              // lds_addr (bytes)
                 (unsigned int)ga,                     // global_addr[31:0]
                 (unsigned int)(ga >> 32) | (2u << 30) // global_addr[56:32] | type=2
               };
    i32x8 g1 = { (1 << 16) | (1 << 20) | (6 << 22) | (3 << 25), // dsz=2B|pad_en|intv|amt
                 (CDIM & 0xFFFF) << 16,   // tensor_dim0 lo16 (abar_addr=0)
                 (KCB  & 0xFFFF) << 16,   // tensor_dim0 hi=0 | tensor_dim1 lo16
                 (CDIM) << 16,            // tensor_dim1 hi=0 | tile_dim0=256
                 KTILE,                   // tile_dim1=32 | tile_dim2=0
                 CDIM,                    // tensor_dim0_stride lo32 = 256
                 0,                       // stride0 hi | tensor_dim1_stride lo16
                 0 };
    i32x4 gz = { 0, 0, 0, 0 };
#if defined(__clang_major__) && (__clang_major__ >= 23)
    i32x8 gz8 = { 0, 0, 0, 0, 0, 0, 0, 0 };
    __builtin_amdgcn_tensor_load_to_lds(g0, g1, gz, gz, gz8, 0);
#else
    __builtin_amdgcn_tensor_load_to_lds(g0, g1, gz, gz, 0);
#endif
}

// ---------------------------------------------------------------------------
// Kernel 2: fused f16-WMMA Gram matrix + argmin, TDM double-buffered B tiles.
// score[n,k] = ||c_k||^2 - 2 * x_n . c_k   (||x_n||^2 row-constant -> dropped)
// 8 waves = 4 row-tiles x 2 col-tiles -> 64x32 scores per K-chunk.
// ---------------------------------------------------------------------------
__global__ __launch_bounds__(256) void vq_argmin_wmma(const _Float16* __restrict__ lat_h,
                                                      const _Float16* __restrict__ cb_h,
                                                      const float* __restrict__ cnorm,
                                                      int* __restrict__ inds_det) {
    extern __shared__ unsigned char smem[];   // SMEM_TOTAL bytes
    _Float16* As = (_Float16*)smem;
    const unsigned int lds_base = (unsigned int)(uintptr_t)(void*)smem;

    const int tid  = threadIdx.x;
    const int lane = tid & 31;          // wave32
    const int wave = tid >> 5;
    const int rt   = wave >> 1;         // row tile 0..3
    const int ct   = wave & 1;          // col tile 0..1
    const int nbase = blockIdx.x * NTILE;

    // Kick off TDM for chunk 0 into B buffer 0 (wave 0 only), then stage A.
    if (tid < 32) tdm_load_tile(cb_h, lds_base + A_BYTES);

    {   // stage A tile (64 rows x 256 halfs) into padded LDS
        const uint4* src = (const uint4*)(lat_h + (size_t)nbase * CDIM);
        for (int i = tid; i < NTILE * 32; i += 256) {   // 32 uint4 per row
            const int row = i >> 5, g = i & 31;
            uint4 v = src[(size_t)row * 32 + g];
            *(uint4*)(As + row * ASTRIDE + g * 8) = v;
        }
    }
    __syncthreads();

    // Hoist the 8 A fragments for this wave's 16-row slice.
    const int arow  = rt * 16 + (lane & 15);
    const int khalf = (lane >> 4) * 16;     // K-half select: lane half h holds k = 16h..16h+15 (same order for B)
    v16h afrag[8];
    #pragma unroll
    for (int cc = 0; cc < 8; ++cc)
        afrag[cc] = ld_frag(As + arow * ASTRIDE + cc * 32 + khalf);

    float minval[8];
    int   minidx[8];
    #pragma unroll
    for (int j = 0; j < 8; ++j) { minval[j] = 3.4e38f; minidx[j] = 0; }

    const int bcol = ct * 16 + (lane & 15); // this lane's column within the chunk

    for (int chunk = 0; chunk < NCHUNK; ++chunk) {
        __syncthreads();   // compute(chunk-1) finished -> other buffer reusable
        if (tid < 32) {
            if (chunk + 1 < NCHUNK) {
                tdm_load_tile(cb_h + (size_t)(chunk + 1) * KTILE * CDIM,
                              lds_base + A_BYTES + ((chunk + 1) & 1) * BBUF_BYTES);
                __builtin_amdgcn_s_wait_tensorcnt(1);  // chunk's tile landed
            } else {
                __builtin_amdgcn_s_wait_tensorcnt(0);
            }
        }
        __syncthreads();   // publish buf[chunk&1] to all waves

        const _Float16* Bs = (const _Float16*)(smem + A_BYTES + (chunk & 1) * BBUF_BYTES);
        v8f acc = {};
        #pragma unroll
        for (int cc = 0; cc < 8; ++cc) {
            v16h bfrag = ld_frag(Bs + bcol * ASTRIDE + cc * 32 + khalf);
            acc = __builtin_amdgcn_wmma_f32_16x16x32_f16(
                      false, afrag[cc], false, bfrag, (short)0, acc, false, false);
        }

        const int   k  = chunk * KTILE + bcol;
        const float cn = cnorm[k];
        #pragma unroll
        for (int j = 0; j < 8; ++j) {
            const float score = fmaf(-2.0f, acc[j], cn);
            if (score < minval[j]) { minval[j] = score; minidx[j] = k; }
        }
    }

    // ---- cross-lane / cross-wave argmin via LDS scratch overlaying dead B0 ----
    __syncthreads();
    float* sval = (float*)(smem + A_BYTES);
    int*   sidx = (int*)(smem + A_BYTES + NTILE * 32 * sizeof(float));
    const int slot = ct * 16 + (lane & 15);
    #pragma unroll
    for (int j = 0; j < 8; ++j) {
        const int row = rt * 16 + j + ((lane >> 4) << 3);  // C/D: M = j + 8*(lane>=16)
        sval[row * 32 + slot] = minval[j];
        sidx[row * 32 + slot] = minidx[j];
    }
    __syncthreads();
    if (tid < NTILE) {
        float bv = sval[tid * 32];
        int   bi = sidx[tid * 32];
        #pragma unroll 4
        for (int s = 1; s < 32; ++s) {
            const float v = sval[tid * 32 + s];
            const int   i2 = sidx[tid * 32 + s];
            if (v < bv || (v == bv && i2 < bi)) { bv = v; bi = i2; }
        }
        inds_det[nbase + tid] = bi;
    }
}

// ---------------------------------------------------------------------------
// Kernel 3: noisy index, gathers, straight-through output, fused loss.
// ---------------------------------------------------------------------------
__global__ __launch_bounds__(256) void vq_finalize(const float* __restrict__ lat,
                                                   const float* __restrict__ cb,
                                                   const float* __restrict__ noise,
                                                   const int* __restrict__ inds_det,
                                                   int* __restrict__ inds_noisy,
                                                   float* __restrict__ out,
                                                   float* __restrict__ loss) {
    __shared__ float ws[8];
    const int n   = blockIdx.x;
    const int tid = threadIdx.x;

    const int id  = inds_det[n];
    int in_ = id + (int)rintf(noise[n]);           // round-half-even == jnp.round
    in_ = in_ < 0 ? 0 : (in_ > (KCB - 1) ? (KCB - 1) : in_);

    const float l  = lat[(size_t)n * CDIM + tid];
    const float qd = cb[(size_t)id  * CDIM + tid];
    const float qn = cb[(size_t)in_ * CDIM + tid];

    out[(size_t)n * CDIM + tid] = qn;              // latents + sg(q_noisy - latents)

    const float e1 = l - qd;
    const float e2 = qn - l;
    float p = BETA * e1 * e1 + e2 * e2;
    #pragma unroll
    for (int off = 16; off > 0; off >>= 1) p += __shfl_xor(p, off, 32);
    if ((tid & 31) == 0) ws[tid >> 5] = p;
    __syncthreads();
    if (tid == 0) {
        float t = 0.0f;
        #pragma unroll
        for (int i = 0; i < 8; ++i) t += ws[i];
        atomicAdd(loss, t * (1.0f / ((float)NTOT * (float)CDIM)));
        inds_noisy[n] = in_;
    }
}

// ---------------------------------------------------------------------------
extern "C" void kernel_launch(void* const* d_in, const int* in_sizes, int n_in,
                              void* d_out, int out_size, void* d_ws, size_t ws_size,
                              hipStream_t stream) {
    const float* lat   = (const float*)d_in[0];   // [32,256,256]
    const float* cb    = (const float*)d_in[1];   // [8192,256]
    const float* noise = (const float*)d_in[2];   // [8192,1]

    float* out        = (float*)d_out;                 // 2,097,152 f32
    float* loss       = out + (size_t)NTOT * CDIM;     // 1 f32
    int*   inds_det   = (int*)(loss + 1);              // 8192 i32
    int*   inds_noisy = inds_det + NTOT;               // 8192 i32

    _Float16* lat_h = (_Float16*)d_ws;                         // 4 MB
    _Float16* cb_h  = lat_h + (size_t)NTOT * CDIM;             // 4 MB
    float*    cnorm = (float*)(cb_h + (size_t)KCB * CDIM);     // 32 KB

    vq_init<<<1, 1, 0, stream>>>(loss);
    vq_prep<<<KCB + NTOT, 256, 0, stream>>>(lat, cb, lat_h, cb_h, cnorm);
    vq_argmin_wmma<<<NTOT / NTILE, 256, SMEM_TOTAL, stream>>>(lat_h, cb_h, cnorm, inds_det);
    vq_finalize<<<NTOT, 256, 0, stream>>>(lat, cb, noise, inds_det, inds_noisy, out, loss);
}
